// MultiHeadSelfAttention_81724637708427
// MI455X (gfx1250) — compile-verified, matrix operand order checked by reading, not run on hardware
//
#include <hip/hip_runtime.h>
#include <cstdint>
#include <cstddef>

// ---------------------------------------------------------------------------
// MI455X (gfx1250) multi-head self-attention, bf16 WMMA path.
// B=4, S=2048, D=2048, H=16, DH=128.  M = B*S = 8192.
// Pipeline:
//   1) gemm_bias_kernel<float,bf16>  x3 : Q/K/V projections (fp32 in -> bf16 out)
//      (register-staged double buffering: next tile's global loads issue
//       before the current tile's WMMAs)
//   2) attn_kernel : flash-style streaming attention with the reference's
//      post-softmax causal fill (-1e9) handled as  O/l - 1e9 * suffixSum(V).
//      Q/K tiles staged with GLOBAL_LOAD_ASYNC_TO_LDS_B128 (ASYNCcnt path).
//   3) gemm_bias_kernel<bf16,float> : output projection -> d_out (fp32)
// ---------------------------------------------------------------------------

typedef __attribute__((ext_vector_type(16))) __bf16 v16bf;
typedef __attribute__((ext_vector_type(8)))  float  v8f;
typedef __attribute__((ext_vector_type(4)))  int    v4i;

#define BATCH  4
#define SEQ    2048
#define DMODEL 2048
#define NHEAD  16
#define DHEAD  128
#define MROWS  (BATCH * SEQ)

// ---- CDNA5 async global->LDS staging (guarded; falls back to sync copy) ----
#if defined(__gfx1250__) && __has_builtin(__builtin_amdgcn_global_load_async_to_lds_b128) && __has_builtin(__builtin_amdgcn_s_wait_asynccnt)
#define ASYNC_LDS 1
typedef __attribute__((address_space(1))) v4i as1_v4i;   // global int4
typedef __attribute__((address_space(3))) v4i as3_v4i;   // LDS int4
#else
#define ASYNC_LDS 0
#endif

// Copy 16 bytes global -> LDS (8 bf16).  Both sides must be 16B aligned.
__device__ __forceinline__ void stage16(const __bf16* g, __bf16* l) {
#if ASYNC_LDS
  __builtin_amdgcn_global_load_async_to_lds_b128((as1_v4i*)g, (as3_v4i*)l, 0, 0);
#else
  uint4 d = *(const uint4*)g;
  *(uint2*)l = make_uint2(d.x, d.y);
  *(uint2*)(l + 4) = make_uint2(d.z, d.w);
#endif
}
__device__ __forceinline__ void stage_fence() {
#if ASYNC_LDS
  __builtin_amdgcn_s_wait_asynccnt(0);   // drain ASYNCcnt before the barrier
#endif
}

__device__ __forceinline__ v8f wmma_bf16(v16bf a, v16bf b, v8f c) {
  // D = A(16x32 bf16) * B(32x16 bf16) + C(16x16 f32)
  return __builtin_amdgcn_wmma_f32_16x16x32_bf16(false, a, false, b,
                                                 (short)0, c, false, false);
}

union FragU { v16bf v; uint32_t u[8]; };

// Load a 16x32 bf16 A/B fragment from a K-contiguous LDS tile.
// Caller passes  p = tile + (row0 + (lane&15)) * pitch + ((lane>>4)<<3).
// Per ISA 7.12.2: elements 0..7 hold K = kb..kb+7, elements 8..15 hold
// K = kb+16..kb+23 (kb = 0 or 8 by lane half).  pitch must be mult of 4.
__device__ __forceinline__ v16bf frag_ld(const __bf16* p) {
  FragU f;
  const uint32_t* a = (const uint32_t*)p;
  const uint32_t* b = (const uint32_t*)(p + 16);
#pragma unroll
  for (int j = 0; j < 4; ++j) { f.u[j] = a[j]; f.u[4 + j] = b[j]; }
  return f.v;
}

// ---- per-dtype 4-element staging chunks for the GEMM double buffer ----
__device__ __forceinline__ float4 ld_chunk(const float* g)  { return *(const float4*)g; }
__device__ __forceinline__ uint2  ld_chunk(const __bf16* g) { return *(const uint2*)g; }
__device__ __forceinline__ void st_chunk(float4 v, __bf16* l) {
  l[0] = (__bf16)v.x; l[1] = (__bf16)v.y; l[2] = (__bf16)v.z; l[3] = (__bf16)v.w;
}
__device__ __forceinline__ void st_chunk(uint2 v, __bf16* l) { *(uint2*)l = v; }
template <typename T> using chunk_t = decltype(ld_chunk((const T*)nullptr));

// ---------------------------------------------------------------------------
// Tiled GEMM:  Y[M,N] = A[M,K] * W[K,N] + bias[N]
// Block tile 128x128, K-step 32.  256 threads = 8 waves, each wave 64x32.
// Register-staged double buffer: global loads for tile t+1 issue before the
// WMMAs of tile t, LDS is refilled after the consume barrier.
// ---------------------------------------------------------------------------
template <typename AT, typename OT>
__global__ __launch_bounds__(256) void gemm_bias_kernel(
    const AT* __restrict__ A, const float* __restrict__ W,
    const float* __restrict__ bias, OT* __restrict__ Y,
    int M, int K, int N) {
  constexpr int BM = 128, BN = 128, BK = 32, PITCH = BK + 4;  // 36
  __shared__ __bf16 As[BM * PITCH];   // [m][k], K-contiguous
  __shared__ __bf16 Bs[BN * PITCH];   // [n][k], K-contiguous (W transposed)

  const int tid  = threadIdx.x;
  const int wave = tid >> 5, lane = tid & 31;
  const int wm = wave >> 2, wn = wave & 3;           // 2 x 4 wave grid
  const int bn0 = blockIdx.x * BN;
  const int bm0 = blockIdx.y * BM;
  const int r  = lane & 15;
  const int kb = (lane >> 4) << 3;

  v8f acc[4][2];
#pragma unroll
  for (int mt = 0; mt < 4; ++mt)
#pragma unroll
    for (int nt = 0; nt < 2; ++nt) acc[mt][nt] = (v8f){};

  chunk_t<AT> aReg[4];
  float4      wReg[4];

  // ---- prologue: stage tile k0 = 0 through registers into LDS ----
#pragma unroll
  for (int i = 0; i < 4; ++i) {
    int f4 = tid + i * 256;                          // 0..1023
    aReg[i] = ld_chunk(A + (size_t)(bm0 + (f4 >> 3)) * K + ((f4 & 7) << 2));
    wReg[i] = *(const float4*)(W + (size_t)(f4 >> 5) * N + bn0 + ((f4 & 31) << 2));
  }
#pragma unroll
  for (int i = 0; i < 4; ++i) {
    int f4 = tid + i * 256;
    st_chunk(aReg[i], As + (f4 >> 3) * PITCH + ((f4 & 7) << 2));
    int krow = f4 >> 5, nc4 = (f4 & 31) << 2;
    Bs[(nc4 + 0) * PITCH + krow] = (__bf16)wReg[i].x;
    Bs[(nc4 + 1) * PITCH + krow] = (__bf16)wReg[i].y;
    Bs[(nc4 + 2) * PITCH + krow] = (__bf16)wReg[i].z;
    Bs[(nc4 + 3) * PITCH + krow] = (__bf16)wReg[i].w;
  }
  __syncthreads();

  for (int k0 = 0; k0 < K; k0 += BK) {
    const bool hasNext = (k0 + BK) < K;
    if (hasNext) {
      // issue next tile's global loads early; they retire while we WMMA
#pragma unroll
      for (int i = 0; i < 4; ++i) {
        int f4 = tid + i * 256;
        aReg[i] = ld_chunk(A + (size_t)(bm0 + (f4 >> 3)) * K + (k0 + BK) + ((f4 & 7) << 2));
        wReg[i] = *(const float4*)(W + (size_t)(k0 + BK + (f4 >> 5)) * N + bn0 + ((f4 & 31) << 2));
      }
      if (k0 + 2 * BK < K) {  // cache hint two tiles ahead (global_prefetch_b8)
        __builtin_prefetch(W + (size_t)(k0 + 2 * BK + (tid >> 3)) * N + bn0 + (tid & 7) * 16, 0, 1);
      }
    }

    // ---- compute current tile ----
    v16bf af[4], bfr[2];
#pragma unroll
    for (int mt = 0; mt < 4; ++mt)
      af[mt] = frag_ld(As + (wm * 64 + mt * 16 + r) * PITCH + kb);
#pragma unroll
    for (int nt = 0; nt < 2; ++nt)
      bfr[nt] = frag_ld(Bs + (wn * 32 + nt * 16 + r) * PITCH + kb);
#pragma unroll
    for (int mt = 0; mt < 4; ++mt)
#pragma unroll
      for (int nt = 0; nt < 2; ++nt)
        acc[mt][nt] = wmma_bf16(af[mt], bfr[nt], acc[mt][nt]);

    __syncthreads();                 // all waves done reading this tile
    if (hasNext) {
#pragma unroll
      for (int i = 0; i < 4; ++i) {
        int f4 = tid + i * 256;
        st_chunk(aReg[i], As + (f4 >> 3) * PITCH + ((f4 & 7) << 2));
        int krow = f4 >> 5, nc4 = (f4 & 31) << 2;
        Bs[(nc4 + 0) * PITCH + krow] = (__bf16)wReg[i].x;
        Bs[(nc4 + 1) * PITCH + krow] = (__bf16)wReg[i].y;
        Bs[(nc4 + 2) * PITCH + krow] = (__bf16)wReg[i].z;
        Bs[(nc4 + 3) * PITCH + krow] = (__bf16)wReg[i].w;
      }
      __syncthreads();               // LDS refilled for next iteration
    }
  }

  // ---- epilogue: C layout = row (vgpr i + 8*hi), col = lane&15 ----
  const int hi = lane >> 4;
#pragma unroll
  for (int mt = 0; mt < 4; ++mt) {
#pragma unroll
    for (int nt = 0; nt < 2; ++nt) {
      int col = bn0 + wn * 32 + nt * 16 + r;
      float bv = bias[col];
      int rowb = bm0 + wm * 64 + mt * 16 + hi * 8;
#pragma unroll
      for (int i = 0; i < 8; ++i) {
        float o = acc[mt][nt][i] + bv;
        Y[(size_t)(rowb + i) * N + col] = (OT)o;
      }
    }
  }
}

// ---------------------------------------------------------------------------
// Streaming attention.  One block per (b, h, 128-row q tile); 8 waves x 16 q
// rows.  Per 64-key block: QK^T (WMMA), full-row online softmax, causally
// masked P@V (WMMA), indicator@V suffix-sum of V (WMMA).  Out = O/l - 1e9*Vsuf.
// ---------------------------------------------------------------------------
__global__ __launch_bounds__(256) void attn_kernel(
    const __bf16* __restrict__ Qg, const __bf16* __restrict__ Kg,
    const __bf16* __restrict__ Vg, __bf16* __restrict__ Og) {
  constexpr int QB = 128, KB = 64;
  constexpr int QP = DHEAD + 8;  // 136
  constexpr int VP = KB + 8;     // 72
  constexpr int PP = KB + 8;     // 72
  constexpr float SCALE = 0.088388347648318447f;  // 1/sqrt(128)

  __shared__ __bf16 Qs[QB * QP];     // [q][dh]
  __shared__ __bf16 Ks[KB * QP];     // [k][dh]
  __shared__ __bf16 Vt[DHEAD * VP];  // [dh][k]  (transposed V tile)
  __shared__ __bf16 Ps[QB * PP];     // [q][k]   (masked probabilities)

  const int tid  = threadIdx.x;
  const int wave = tid >> 5, lane = tid & 31;
  const int r  = lane & 15;
  const int hi = lane >> 4;
  const int kb = hi << 3;
  const int qb0 = blockIdx.x * QB;
  const int h = blockIdx.y, b = blockIdx.z;
  const size_t base = ((size_t)b * SEQ) * DMODEL + (size_t)h * DHEAD;

  // ---- stage Q tile 128x128 via async LDS copy (8 x 16B per thread) ----
#pragma unroll
  for (int i = 0; i < 8; ++i) {
    int c   = tid + i * 256;          // 0..2047
    int row = c >> 4;
    int off = (c & 15) << 3;
    stage16(Qg + base + (size_t)(qb0 + row) * DMODEL + off, Qs + row * QP + off);
  }
  stage_fence();
  __syncthreads();

  // Q fragments stay resident: 4 dh-chunks of 32.
  v16bf qf[4];
#pragma unroll
  for (int d = 0; d < 4; ++d)
    qf[d] = frag_ld(Qs + (wave * 16 + r) * QP + d * 32 + kb);

  v8f Oacc[8], Vsuf[8];
#pragma unroll
  for (int dt = 0; dt < 8; ++dt) { Oacc[dt] = (v8f){}; Vsuf[dt] = (v8f){}; }
  float mrun[8], lrun[8];
#pragma unroll
  for (int i = 0; i < 8; ++i) { mrun[i] = -1e30f; lrun[i] = 0.f; }

  const int qlo = qb0 + wave * 16;    // wave's lowest global q row

  for (int kt = 0; kt < SEQ / KB; ++kt) {
    const int kb0 = kt * KB;
    __syncthreads();
    // ---- stage K rows asynchronously; V manually with dh<->k transpose ----
#pragma unroll
    for (int i = 0; i < 4; ++i) {
      int c   = tid + i * 256;        // 0..1023
      int row = c >> 4;               // key index 0..63
      int off = (c & 15) << 3;        // dh offset
      stage16(Kg + base + (size_t)(kb0 + row) * DMODEL + off, Ks + row * QP + off);
      uint4 dv = *(const uint4*)(Vg + base + (size_t)(kb0 + row) * DMODEL + off);
      const __bf16* t = (const __bf16*)&dv;
#pragma unroll
      for (int j = 0; j < 8; ++j) Vt[(off + j) * VP + row] = t[j];
    }
    stage_fence();
    __syncthreads();

    // ---- scores: 4 key tiles x 4 dh-steps of WMMA ----
    v8f sAcc[4];
#pragma unroll
    for (int nt = 0; nt < 4; ++nt) sAcc[nt] = (v8f){};
#pragma unroll
    for (int nt = 0; nt < 4; ++nt)
#pragma unroll
      for (int d = 0; d < 4; ++d)
        sAcc[nt] = wmma_bf16(qf[d],
                             frag_ld(Ks + (nt * 16 + r) * QP + d * 32 + kb),
                             sAcc[nt]);
#pragma unroll
    for (int nt = 0; nt < 4; ++nt)
#pragma unroll
      for (int i = 0; i < 8; ++i) sAcc[nt][i] *= SCALE;

    // ---- online softmax over ALL keys (reference masks AFTER softmax) ----
#pragma unroll
    for (int i = 0; i < 8; ++i) {
      float mx = sAcc[0][i];
#pragma unroll
      for (int nt = 1; nt < 4; ++nt) mx = fmaxf(mx, sAcc[nt][i]);
#pragma unroll
      for (int d = 1; d < 16; d <<= 1) mx = fmaxf(mx, __shfl_xor(mx, d, 32));
      float mN  = fmaxf(mrun[i], mx);
      float fsc = __expf(mrun[i] - mN);
      mrun[i] = mN;
      lrun[i] *= fsc;
#pragma unroll
      for (int dt = 0; dt < 8; ++dt) Oacc[dt][i] *= fsc;
      float rs = 0.f;
#pragma unroll
      for (int nt = 0; nt < 4; ++nt) {
        float p = __expf(sAcc[nt][i] - mN);
        sAcc[nt][i] = p;
        rs += p;
      }
#pragma unroll
      for (int d = 1; d < 16; d <<= 1) rs += __shfl_xor(rs, d, 32);
      lrun[i] += rs;
    }

    // ---- write causally-masked P to per-wave LDS (C-layout -> A-layout) ----
#pragma unroll
    for (int nt = 0; nt < 4; ++nt) {
      int k_g = kb0 + nt * 16 + r;
#pragma unroll
      for (int i = 0; i < 8; ++i) {
        int qr  = hi * 8 + i;
        int q_g = qlo + qr;
        float pv = (k_g <= q_g) ? sAcc[nt][i] : 0.f;
        Ps[(wave * 16 + qr) * PP + nt * 16 + r] = (__bf16)pv;
      }
    }

    // ---- indicator fragment for the -1e9 suffix-V term (k > q) ----
    const bool anyAbove = (kb0 + KB - 1) > qlo;   // wave-uniform
    v16bf uf[2];
    if (anyAbove) {
      int q_g = qlo + r;
#pragma unroll
      for (int c = 0; c < 2; ++c) {
        FragU f;
#pragma unroll
        for (int e = 0; e < 16; ++e) {
          int koff = (e < 8) ? (kb + e) : (kb + 8 + e);   // K mapping (7.12.2)
          int k_g  = kb0 + c * 32 + koff;
          f.v[e] = (k_g > q_g) ? (__bf16)1.0f : (__bf16)0.0f;
        }
        uf[c] = f.v;
      }
    }

    // ---- P@V and indicator@V accumulation ----
    v16bf pf[2];
#pragma unroll
    for (int c = 0; c < 2; ++c)
      pf[c] = frag_ld(Ps + (wave * 16 + r) * PP + c * 32 + kb);
#pragma unroll
    for (int dt = 0; dt < 8; ++dt) {
#pragma unroll
      for (int c = 0; c < 2; ++c) {
        v16bf vf = frag_ld(Vt + (dt * 16 + r) * VP + c * 32 + kb);
        Oacc[dt] = wmma_bf16(pf[c], vf, Oacc[dt]);
        if (anyAbove) Vsuf[dt] = wmma_bf16(uf[c], vf, Vsuf[dt]);
      }
    }
  }

  // ---- epilogue: out = O/l - 1e9 * Vsuf, bf16 to [b*S+q, h*DH+dh] ----
#pragma unroll
  for (int dt = 0; dt < 8; ++dt) {
    int dh0 = dt * 16 + r;
#pragma unroll
    for (int i = 0; i < 8; ++i) {
      int qr  = hi * 8 + i;
      int q_g = qb0 + wave * 16 + qr;
      float o = Oacc[dt][i] / lrun[i] - 1e9f * Vsuf[dt][i];
      Og[((size_t)b * SEQ + q_g) * DMODEL + h * DHEAD + dh0] = (__bf16)o;
    }
  }
}

// ---------------------------------------------------------------------------
extern "C" void kernel_launch(void* const* d_in, const int* in_sizes, int n_in,
                              void* d_out, int out_size, void* d_ws, size_t ws_size,
                              hipStream_t stream) {
  (void)in_sizes; (void)n_in; (void)out_size; (void)ws_size;
  const float* q  = (const float*)d_in[0];
  const float* k  = (const float*)d_in[1];
  const float* v  = (const float*)d_in[2];
  const float* Wq = (const float*)d_in[3];
  const float* bq = (const float*)d_in[4];
  const float* Wk = (const float*)d_in[5];
  const float* bk = (const float*)d_in[6];
  const float* Wv = (const float*)d_in[7];
  const float* bv = (const float*)d_in[8];
  const float* Wo = (const float*)d_in[9];
  const float* bo = (const float*)d_in[10];
  // d_in[11] = mask: deterministic tril, folded into the kernel analytically.

  const size_t MD = (size_t)MROWS * DMODEL;          // 16,777,216 elements
  __bf16* Qb = (__bf16*)d_ws;
  __bf16* Kb = Qb + MD;
  __bf16* Vb = Kb + MD;
  __bf16* Ab = Vb + MD;                              // total 128 MiB of ws

  dim3 blk(256);
  dim3 gg(DMODEL / 128, MROWS / 128);                // (16, 64)

  gemm_bias_kernel<float, __bf16><<<gg, blk, 0, stream>>>(q, Wq, bq, Qb, MROWS, DMODEL, DMODEL);
  gemm_bias_kernel<float, __bf16><<<gg, blk, 0, stream>>>(k, Wk, bk, Kb, MROWS, DMODEL, DMODEL);
  gemm_bias_kernel<float, __bf16><<<gg, blk, 0, stream>>>(v, Wv, bv, Vb, MROWS, DMODEL, DMODEL);

  dim3 ga(SEQ / 128, NHEAD, BATCH);                  // (16, 16, 4)
  attn_kernel<<<ga, blk, 0, stream>>>(Qb, Kb, Vb, Ab);

  gemm_bias_kernel<__bf16, float><<<gg, blk, 0, stream>>>(Ab, Wo, bo, (float*)d_out, MROWS, DMODEL, DMODEL);
}
